// Quant3Linear_fused_eora_dummy_32762010534271
// MI455X (gfx1250) — matrix operand order repaired and checked by reading, not compiled, not run on hardware
//
#include <hip/hip_runtime.h>
#include <hip/hip_bf16.h>
#include <stdint.h>

// ---------------------------------------------------------------------------
// GPTQ 3-bit GEMV (1x8192 @ 8192x8192) + EoRA low-rank, fused on gfx1250.
// Bandwidth-bound (~29.5 MB @ 23.3 TB/s => ~1.3us floor). MACs go through
// v_wmma_f32_16x16x32_f16; VALU only does 3-bit unpack via the 0x4800|(q<<7)
// == (8+q) f16 magic-constant trick, corrected by 8*S in the epilogue.
// Loop unrolled x2 so the scheduler can pipeline global/LDS loads + unpack
// of one chunk under the other's WMMA (round-1 asm showed s_wait_dscnt 0
// right before the single WMMA, i.e. no cross-iteration overlap).
// ---------------------------------------------------------------------------

typedef __attribute__((ext_vector_type(16))) _Float16 v16h;
typedef __attribute__((ext_vector_type(8)))  float    v8f;
typedef __attribute__((ext_vector_type(8)))  unsigned v8u;
typedef __attribute__((ext_vector_type(4)))  unsigned v4u;

#define IN_FEATURES  8192
#define OUT_FEATURES 8192
#define RANK         64
#define KSPLIT       4
#define GROUPS       (IN_FEATURES / 32)          // 256 groups of 32 codes
#define GROUPS_PER_SPLIT (GROUPS / KSPLIT)       // 64
#define K_PER_SPLIT  (IN_FEATURES / KSPLIT)      // 2048

// ws layout (floats): [0..63] t = x@down, [64] S = sum(x), [128 ..] KSPLIT*8192 partial dsums

// ---------------------------------------------------------------------------
// Kernel 1: t[r] = dot(x, down[:,r]) for r<64 ; block 64 computes S = sum(x)
// ---------------------------------------------------------------------------
__global__ __launch_bounds__(256) void k_lowrank(const float* __restrict__ x,
                                                 const float* __restrict__ down,
                                                 float* __restrict__ ws) {
  __shared__ float red[256];
  const int r = blockIdx.x;
  const int t = threadIdx.x;
  float acc = 0.f;
  if (r < RANK) {
    for (int k = t; k < IN_FEATURES; k += 256) acc += x[k] * down[k * RANK + r];
  } else {
    for (int k = t; k < IN_FEATURES; k += 256) acc += x[k];
  }
  red[t] = acc;
  __syncthreads();
  for (int s = 128; s > 0; s >>= 1) {
    if (t < s) red[t] += red[t + s];
    __syncthreads();
  }
  if (t == 0) ws[r] = red[0];
}

// ---------------------------------------------------------------------------
// Kernel 2: WMMA main GEMV with split-K.
//  grid = 256 blocks x 256 threads. block -> (tg = bx>>2, ksplit = bx&3).
//  Each wave owns 16 output channels for one K-split (64 groups of 32 k).
//  A operand: x (f16, from LDS) broadcast to all 16 rows.
//  B operand: dequant-coded weights, f16 value (8+q) built by bit ops.
//  Partial D (= 8*S_split + dot(x_split, q)) -> ws partials, combined later.
// ---------------------------------------------------------------------------
__global__ __launch_bounds__(256) void k_wmma(const unsigned* __restrict__ qw,
                                              const float* __restrict__ x,
                                              float* __restrict__ part) {
  __shared__ __align__(16) _Float16 xs[K_PER_SPLIT];   // 4 KB
  const int tid  = threadIdx.x;
  const int ks   = blockIdx.x & (KSPLIT - 1);
  const int tg   = blockIdx.x >> 2;
  const int kbase = ks * K_PER_SPLIT;

  // cooperative f32 -> f16 conversion of this split's slice of x
  for (int i = tid; i < K_PER_SPLIT; i += 256) xs[i] = (_Float16)x[kbase + i];
  __syncthreads();

  const int wave = tid >> 5;
  const int lane = tid & 31;
  const int hf   = lane >> 4;      // which 16-lane half
  const int col  = lane & 15;      // output channel within tile / B column N
  const int o    = (tg * 8 + wave) * 16 + col;

  v8f c = {0.f, 0.f, 0.f, 0.f, 0.f, 0.f, 0.f, 0.f};
  const int g0 = ks * GROUPS_PER_SPLIT;

#pragma unroll 2
  for (int gl = 0; gl < GROUPS_PER_SPLIT; ++gl) {
    const int g = g0 + gl;
    // B-operand: lane (hf,col) needs codes [16*hf .. 16*hf+15] of the 96-bit
    // group = a 48-bit window starting at bit 48*hf -> words {hf, hf+1}.
    // Coalesced: 16 consecutive o per half-wave = one 64B line per load.
    const unsigned wlo = qw[(g * 3 + hf) * OUT_FEATURES + o];
    const unsigned whi = qw[(g * 3 + hf + 1) * OUT_FEATURES + o];
    // software prefetch next group's stream (96KB stride per wave)
    {
      const int gp = (gl + 1 < GROUPS_PER_SPLIT) ? (g + 1) : g;
      __builtin_prefetch(&qw[(gp * 3 + hf) * OUT_FEATURES + o], 0, 0);
    }
    const unsigned long long W =
        ((((unsigned long long)whi << 32) | wlo) >> (hf ? 16 : 0));

    // Build B: VGPR r packs f16 codes (2r, 2r+1): f16 bits 0x4800|(q<<7) == 8+q
    v8u bu;
#pragma unroll
    for (int r = 0; r < 8; ++r) {
      const unsigned t2 = (unsigned)(W >> (6 * r));
      bu[r] = 0x48004800u | ((t2 & 7u) << 7) | (((t2 >> 3) & 7u) << 23);
    }

    // A: x broadcast. Lane-half K sets (contiguous, matching B):
    // lanes 0-15: K {0..15}; lanes 16-31: K {16..31}
    const int kl = gl * 32 + hf * 16;
    const v4u alo = *(const v4u*)(&xs[kl]);        // ds_load_b128
    const v4u ahi = *(const v4u*)(&xs[kl + 8]);    // ds_load_b128
    const v8u au = {alo[0], alo[1], alo[2], alo[3], ahi[0], ahi[1], ahi[2], ahi[3]};

    const v16h a = __builtin_bit_cast(v16h, au);
    const v16h b = __builtin_bit_cast(v16h, bu);
    c = __builtin_amdgcn_wmma_f32_16x16x32_f16(
        /*neg_a=*/false, a, /*neg_b=*/false, b,
        /*c_mod=*/(short)0, c, /*reuse_a=*/false, /*reuse_b=*/false);
  }

  // All D rows identical (A rows identical): c[0] holds the value for output
  // channel o on every lane; half 0 writes the partial.
  if (hf == 0) part[ks * OUT_FEATURES + o] = c[0];
}

// ---------------------------------------------------------------------------
// Kernel 3: epilogue. y[o] = s_o*(dsum - 8S) - z_o*S + bias + sum_r t[r]*up[r,o]
// ---------------------------------------------------------------------------
__global__ __launch_bounds__(256) void k_final(const float* __restrict__ ws,
                                               const float* __restrict__ scales,
                                               const float* __restrict__ zeros,
                                               const float* __restrict__ bias,
                                               const float* __restrict__ up,
                                               float* __restrict__ out) {
  const int o = blockIdx.x * 256 + threadIdx.x;
  const float S = ws[RANK];
  const float* part = ws + 128;
  float dsum = 0.f;
#pragma unroll
  for (int s = 0; s < KSPLIT; ++s) dsum += part[s * OUT_FEATURES + o];
  float y = scales[o] * (dsum - 8.f * S) - zeros[o] * S + bias[o];
#pragma unroll 8
  for (int r = 0; r < RANK; ++r) y += ws[r] * up[r * OUT_FEATURES + o];
  out[o] = y;
}

// ---------------------------------------------------------------------------
extern "C" void kernel_launch(void* const* d_in, const int* in_sizes, int n_in,
                              void* d_out, int out_size, void* d_ws, size_t ws_size,
                              hipStream_t stream) {
  const float*    x       = (const float*)d_in[0];
  const unsigned* qweight = (const unsigned*)d_in[1];
  const float*    scales  = (const float*)d_in[2];
  const float*    zeros   = (const float*)d_in[3];
  const float*    bias    = (const float*)d_in[4];
  const float*    down    = (const float*)d_in[5];
  const float*    up      = (const float*)d_in[6];
  float*          out     = (float*)d_out;
  float*          ws      = (float*)d_ws;   // needs (128 + KSPLIT*8192)*4 B ~= 132 KB

  k_lowrank<<<RANK + 1, 256, 0, stream>>>(x, down, ws);
  k_wmma<<<(OUT_FEATURES / 16 / 8) * KSPLIT, 256, 0, stream>>>(qweight, x, ws + 128);
  k_final<<<OUT_FEATURES / 256, 256, 0, stream>>>(ws, scales, zeros, bias, up, out);
}
